// MultiScaleRetention_66915590472460
// MI455X (gfx1250) — compile-verified, matrix operand order checked by reading, not run on hardware
//
#include <hip/hip_runtime.h>
#include <hip/hip_bf16.h>

// ---------------------------------------------------------------------------
// Types for CDNA5 WMMA (wave32)
// ---------------------------------------------------------------------------
typedef __attribute__((ext_vector_type(16))) __bf16 v16bf;
typedef __attribute__((ext_vector_type(8)))  __bf16 v8bf;
typedef __attribute__((ext_vector_type(8)))  float  v8f;
typedef __attribute__((ext_vector_type(4)))  unsigned int u32x4;
typedef __attribute__((ext_vector_type(8)))  int i32x8;
typedef __attribute__((ext_vector_type(4)))  int i32x4;

#define HIDDEN 2048
#define HEADS 8
#define HEAD_DIM 256
#define SEQ 4096
#define BATCH 2
#define BN (BATCH * SEQ)   // 8192

static __device__ __forceinline__ __bf16 f2bf(float f) {
    unsigned int u = __float_as_uint(f);
    u += 0x7FFFu + ((u >> 16) & 1u);            // round-to-nearest-even
    union { unsigned short s; __bf16 b; } c;
    c.s = (unsigned short)(u >> 16);
    return c.b;
}

static __device__ __forceinline__ v8f wmma_bf16(v16bf a, v16bf b, v8f c) {
    return __builtin_amdgcn_wmma_f32_16x16x32_bf16(
        false, a, false, b, (short)0, c, false, false);
}

// Fragment loader (A-style for both operands; B tiles are pre-transposed):
// two contiguous 16-byte ds_load_b128 per fragment.
template <int LDSTRIDE>
static __device__ __forceinline__ v16bf load_frag_row(const __bf16* lds, int m0, int k0, int lane) {
    int m  = m0 + (lane & 15);
    int kb = (lane < 16) ? 0 : 8;
    const __bf16* p = lds + m * LDSTRIDE + k0 + kb;
    v8bf lo = *(const v8bf*)(p);
    v8bf hi = *(const v8bf*)(p + 16);
    return __builtin_shufflevector(lo, hi, 0, 1, 2, 3, 4, 5, 6, 7,
                                   8, 9, 10, 11, 12, 13, 14, 15);
}

// ---------------------------------------------------------------------------
// Tensor Data Mover: async 2D bf16 tile global -> LDS with row padding so
// LDS rows land conflict-free (D# per ISA ch.8; pad fields in group1[31:20]).
// ---------------------------------------------------------------------------
#if __has_builtin(__builtin_amdgcn_tensor_load_to_lds)
#define HAVE_TDM 1
#else
#define HAVE_TDM 0
#endif

#if HAVE_TDM
static __device__ __forceinline__ void tdm_load_2d_bf16(
    unsigned int lds_off, const void* gptr,
    unsigned int tile_w, unsigned int tile_h, unsigned int row_stride_elems,
    unsigned int pad_interval, unsigned int pad_amount) {
    unsigned long long ga = (unsigned long long)(uintptr_t)gptr;
    u32x4 g0;
    g0[0] = 1u;                                   // count=1 (valid), user mode
    g0[1] = lds_off;                              // lds_addr (bytes)
    g0[2] = (unsigned int)ga;                     // global_addr[31:0]
    g0[3] = (unsigned int)((ga >> 32) & 0x01FFFFFFu) | (2u << 30); // [56:32]|type=2
    i32x8 g1;
    g1[0] = (int)((1u << 16) | (1u << 20) |       // data_size=2B, pad_enable
                  (pad_interval << 22) | (pad_amount << 25));
    g1[1] = (int)(tile_w << 16);                  // tensor_dim0[15:0]
    g1[2] = (int)((tile_w >> 16) | ((SEQ & 0xFFFFu) << 16)); // td0 hi | td1 lo
    g1[3] = (int)((SEQ >> 16) | (tile_w << 16));  // td1 hi | tile_dim0
    g1[4] = (int)tile_h;                          // tile_dim1 | tile_dim2=0
    g1[5] = (int)row_stride_elems;                // tensor_dim0_stride[31:0]
    g1[6] = 0;
    g1[7] = 0;
    i32x4 z4 = {};
#if __clang_major__ >= 23
    i32x8 z8 = {};
    __builtin_amdgcn_tensor_load_to_lds(g0, g1, z4, z4, z8, 0);
#else
    __builtin_amdgcn_tensor_load_to_lds(g0, g1, z4, z4, 0);
#endif
}
template <int N>
static __device__ __forceinline__ void tdm_wait() {
#if __has_builtin(__builtin_amdgcn_s_wait_tensorcnt)
    __builtin_amdgcn_s_wait_tensorcnt(N);
#else
    asm volatile("s_wait_tensorcnt %0" :: "i"(N) : "memory");
#endif
}
#endif

// ---------------------------------------------------------------------------
// fp32 -> bf16 convert (grid-stride)
// ---------------------------------------------------------------------------
__global__ void cvt_f32_bf16_kernel(const float* __restrict__ src,
                                    __bf16* __restrict__ dst, size_t n) {
    size_t i = (size_t)blockIdx.x * blockDim.x + threadIdx.x;
    size_t stride = (size_t)gridDim.x * blockDim.x;
    for (; i < n; i += stride) dst[i] = f2bf(src[i]);
}

// ---------------------------------------------------------------------------
// Weight transpose+convert: w[K][N] f32 -> Wt[N][K] bf16  (one-time cost, so
// the GEMM hot loop never transposes; 64x64 LDS tiles).
// ---------------------------------------------------------------------------
__global__ __launch_bounds__(256) void transpose_w_kernel(
    const float* __restrict__ src, __bf16* __restrict__ dst) {
    __shared__ __bf16 t[64 * 72];                 // [n][k]
    const int tid = threadIdx.x;
    const int k0 = (blockIdx.x & 31) * 64;
    const int n0 = (blockIdx.x >> 5) * 64;
#pragma unroll
    for (int ii = 0; ii < 4; ++ii) {
        int i = tid + ii * 256;
        int r = i >> 4, c = (i & 15) * 4;         // k row, n cols
        float4 d = *(const float4*)&src[(size_t)(k0 + r) * HIDDEN + n0 + c];
        t[(c + 0) * 72 + r] = f2bf(d.x);
        t[(c + 1) * 72 + r] = f2bf(d.y);
        t[(c + 2) * 72 + r] = f2bf(d.z);
        t[(c + 3) * 72 + r] = f2bf(d.w);
    }
    __syncthreads();
#pragma unroll
    for (int ii = 0; ii < 2; ++ii) {
        int i = tid + ii * 256;
        int r = i >> 3, c = (i & 7) * 8;          // n row, k cols
        *(uint4*)&dst[(size_t)(n0 + r) * HIDDEN + k0 + c] = *(const uint4*)&t[r * 72 + c];
    }
}

// ---------------------------------------------------------------------------
// V transpose+convert: V32[(b*SEQ+n)][HIDDEN] f32 -> VT[(bh*256+d)][SEQ] bf16
// ---------------------------------------------------------------------------
__global__ __launch_bounds__(256) void transpose_v_kernel(
    const float* __restrict__ src, __bf16* __restrict__ dst) {
    __shared__ __bf16 t[64 * 72];                 // [d][n]
    const int tid = threadIdx.x;
    const int n0 = blockIdx.x * 64;
    const int d0 = blockIdx.y * 64;
    const int bh = blockIdx.z;
    const int b = bh >> 3, h = bh & 7;
    const float* sp = src + ((size_t)b * SEQ) * HIDDEN + (size_t)h * HEAD_DIM;
#pragma unroll
    for (int ii = 0; ii < 4; ++ii) {
        int i = tid + ii * 256;
        int r = i >> 4, c = (i & 15) * 4;         // n row, d cols
        float4 d = *(const float4*)&sp[(size_t)(n0 + r) * HIDDEN + d0 + c];
        t[(c + 0) * 72 + r] = f2bf(d.x);
        t[(c + 1) * 72 + r] = f2bf(d.y);
        t[(c + 2) * 72 + r] = f2bf(d.z);
        t[(c + 3) * 72 + r] = f2bf(d.w);
    }
    __syncthreads();
#pragma unroll
    for (int ii = 0; ii < 2; ++ii) {
        int i = tid + ii * 256;
        int r = i >> 3, c = (i & 7) * 8;          // d row, n cols
        *(uint4*)&dst[((size_t)bh * HEAD_DIM + d0 + r) * SEQ + n0 + c] =
            *(const uint4*)&t[r * 72 + c];
    }
}

// ---------------------------------------------------------------------------
// Rotary (theta) shift + scale, fp32 -> bf16.
// ---------------------------------------------------------------------------
__global__ void theta_shift_kernel(const float* __restrict__ src,
                                   __bf16* __restrict__ dst, float scale) {
    int idx = blockIdx.x * blockDim.x + threadIdx.x;
    int row = idx >> 10;
    if (row >= BN) return;
    int p = idx & 1023;
    int n = row & (SEQ - 1);
    int dpair = p & 127;
    float theta = __powf(10000.0f, -(float)dpair * (1.0f / 127.0f));
    float ang = (float)n * theta;
    float s = sinf(ang), c = cosf(ang);
    size_t off = (size_t)row * HIDDEN + 2 * p;
    float xe = src[off], xo = src[off + 1];
    dst[off]     = f2bf((xe * c - xo * s) * scale);
    dst[off + 1] = f2bf((xo * c + xe * s) * scale);
}

// ---------------------------------------------------------------------------
// bf16 WMMA GEMM: C[M,N] = A[M,K] @ Wt[N,K]^T + bias  (A,Wt bf16; C f32)
// Both tiles DMA'd by the Tensor Data Mover into double-buffered LDS (padded
// rows, stride 72); waves do nothing but ds_load_b128 -> v_wmma.
// ---------------------------------------------------------------------------
#define GTS 72    // LDS tile stride (64 + 4-DWORD TDM pad)

__global__ __launch_bounds__(256) void gemm_bf16_kernel(
    const __bf16* __restrict__ A, const __bf16* __restrict__ Wt,
    const float* __restrict__ bias, float* __restrict__ C,
    int M, int K, int N) {
    __shared__ __bf16 ldsA[2][128 * GTS];    // [m][k]
    __shared__ __bf16 ldsW[2][128 * GTS];    // [n][k]

    const int tid = threadIdx.x;
    const int lane = tid & 31;
    const int wave = tid >> 5;
    const int wave_m = wave >> 1;
    const int wave_n = wave & 1;
    const int n0 = blockIdx.x * 128;
    const int m0 = blockIdx.y * 128;

    v8f acc[2][4] = {};

#if HAVE_TDM
    if (wave == 0) {
        tdm_load_2d_bf16((unsigned int)(uintptr_t)&ldsA[0][0],
                         &A[(size_t)m0 * K], 64, 128, K, 4, 3);
        tdm_load_2d_bf16((unsigned int)(uintptr_t)&ldsW[0][0],
                         &Wt[(size_t)n0 * K], 64, 128, K, 4, 3);
    }
#endif
    int buf = 0;
    for (int kt = 0; kt < K; kt += 64) {
        const bool havenext = (kt + 64 < K);
#if HAVE_TDM
        if (wave == 0) {
            if (havenext) {
                tdm_load_2d_bf16((unsigned int)(uintptr_t)&ldsA[buf ^ 1][0],
                                 &A[(size_t)m0 * K + kt + 64], 64, 128, K, 4, 3);
                tdm_load_2d_bf16((unsigned int)(uintptr_t)&ldsW[buf ^ 1][0],
                                 &Wt[(size_t)n0 * K + kt + 64], 64, 128, K, 4, 3);
                tdm_wait<2>();     // current tiles (older pair) complete
            } else {
                tdm_wait<0>();
            }
        }
#else
        for (int i = tid; i < 128 * 8; i += 256) {
            int r = i >> 3, c = (i & 7) * 8;
            *(uint4*)&ldsA[buf][r * GTS + c] =
                *(const uint4*)&A[(size_t)(m0 + r) * K + kt + c];
            *(uint4*)&ldsW[buf][r * GTS + c] =
                *(const uint4*)&Wt[(size_t)(n0 + r) * K + kt + c];
        }
#endif
        __syncthreads();
        const __bf16* bufA = ldsA[buf];
        const __bf16* bufW = ldsW[buf];
#pragma unroll
        for (int kk = 0; kk < 64; kk += 32) {
            v16bf af[2];
#pragma unroll
            for (int mi = 0; mi < 2; ++mi)
                af[mi] = load_frag_row<GTS>(bufA, wave_m * 32 + mi * 16, kk, lane);
#pragma unroll
            for (int ni = 0; ni < 4; ++ni) {
                v16bf bf = load_frag_row<GTS>(bufW, wave_n * 64 + ni * 16, kk, lane);
#pragma unroll
                for (int mi = 0; mi < 2; ++mi)
                    acc[mi][ni] = wmma_bf16(af[mi], bf, acc[mi][ni]);
            }
        }
        __syncthreads();     // all waves done reading buf before next DMA lands
        buf ^= 1;
    }
#pragma unroll
    for (int mi = 0; mi < 2; ++mi)
#pragma unroll
        for (int ni = 0; ni < 4; ++ni) {
            int ncol = n0 + wave_n * 64 + ni * 16 + (lane & 15);
            float b = bias[ncol];
#pragma unroll
            for (int r = 0; r < 8; ++r) {
                int mrow = m0 + wave_m * 32 + mi * 16 + r + ((lane < 16) ? 0 : 8);
                C[(size_t)mrow * N + ncol] = acc[mi][ni][r] + b;
            }
        }
}

// ---------------------------------------------------------------------------
// Retention core.  K tiles (row-major) and V^T tiles both stream through the
// TDM: V(st) and K(st+1) are issued together; in-order TENSORcnt waits let
// S=Q*K^T run while V is still landing, and K(st+1) stays in flight always.
// ---------------------------------------------------------------------------
#define RQS 264   // Q tile stride (256 + 8)
#define RKS 264   // K tile stride (256 + 8, via TDM pad)
#define RVS 72    // V^T tile stride (64 + 8, via TDM pad)
#define RSS 72    // S tile stride (64 + 8)

__global__ __launch_bounds__(256) void retention_kernel(
    const __bf16* __restrict__ QB, const __bf16* __restrict__ KB,
    const __bf16* __restrict__ VT, float* __restrict__ RET) {
    __shared__ __bf16 ldsQ[64 * RQS];       // [n][d]
    __shared__ __bf16 ldsK[2][64 * RKS];    // [s][d], double buffered
    __shared__ __bf16 ldsVt[256 * RVS];     // [d][s]
    __shared__ __bf16 ldsS[64 * RSS];       // [n][s]

    const int tid = threadIdx.x;
    const int lane = tid & 31;
    const int wave = tid >> 5;
    const int n0 = blockIdx.x * 64;
    const int bh = blockIdx.y;
    const int b = bh >> 3, h = bh & 7;
    const size_t base = ((size_t)b * SEQ) * HIDDEN + (size_t)h * HEAD_DIM;
    const size_t vrow0 = (size_t)bh * HEAD_DIM;     // first row of this head in VT

    const float glo = logf(1.0f / 32.0f), ghi = logf(1.0f / 512.0f);
    const float gamma = 1.0f - __expf(glo + (ghi - glo) * (float)h * (1.0f / 7.0f));
    const float lg = logf(gamma);

    const int nTiles = blockIdx.x + 1;

    // resident Q tile
    for (int i = tid; i < 64 * 32; i += 256) {
        int r = i >> 5, c = (i & 31) * 8;
        *(uint4*)&ldsQ[r * RQS + c] =
            *(const uint4*)&QB[base + (size_t)(n0 + r) * HIDDEN + c];
    }
#if HAVE_TDM
    if (wave == 0) {   // kick off DMA of the first K tile
        tdm_load_2d_bf16((unsigned int)(uintptr_t)&ldsK[0][0],
                         &KB[base], HEAD_DIM, 64, HIDDEN, 6, 3);
    }
#endif

    v8f acc[4][2] = {};
    const int mi_s = wave >> 1;
    const int jbase = (wave & 1) * 2;

    int kbuf = 0;
    for (int st = 0; st < nTiles; ++st) {
        const int s0 = st * 64;
        __syncthreads();   // prior iteration finished ldsVt/ldsS and ldsK[kbuf^1]

#if HAVE_TDM
        if (wave == 0) {
            // V tile for THIS iteration (consumed only in phase 2)
            tdm_load_2d_bf16((unsigned int)(uintptr_t)&ldsVt[0],
                             &VT[vrow0 * SEQ + s0], 64, HEAD_DIM, SEQ, 4, 3);
            if (st + 1 < nTiles) {   // K tile for NEXT iteration
                tdm_load_2d_bf16((unsigned int)(uintptr_t)&ldsK[kbuf ^ 1][0],
                                 &KB[base + (size_t)(s0 + 64) * HIDDEN],
                                 HEAD_DIM, 64, HIDDEN, 6, 3);
                tdm_wait<2>();       // K(st) (oldest) complete
            } else {
                tdm_wait<1>();       // only V(st) newer than K(st)
            }
        }
#else
        for (int i = tid; i < 64 * 32; i += 256) {
            int r = i >> 5, c = (i & 31) * 8;
            *(uint4*)&ldsK[kbuf][r * RKS + c] =
                *(const uint4*)&KB[base + (size_t)(s0 + r) * HIDDEN + c];
        }
        for (int i = tid; i < 256 * 8; i += 256) {
            int r = i >> 3, c = (i & 7) * 8;
            *(uint4*)&ldsVt[r * RVS + c] =
                *(const uint4*)&VT[(vrow0 + r) * SEQ + s0 + c];
        }
#endif
        __syncthreads();

        // S = Q * K^T : shared Q fragment, two independent accumulator chains
        const __bf16* kcur = ldsK[kbuf];
        v8f sacc[2] = {};
#pragma unroll
        for (int kk = 0; kk < HEAD_DIM; kk += 32) {
            v16bf qa = load_frag_row<RQS>(ldsQ, mi_s * 16, kk, lane);
            v16bf k0 = load_frag_row<RKS>(kcur, (jbase + 0) * 16, kk, lane);
            v16bf k1 = load_frag_row<RKS>(kcur, (jbase + 1) * 16, kk, lane);
            sacc[0] = wmma_bf16(qa, k0, sacc[0]);
            sacc[1] = wmma_bf16(qa, k1, sacc[1]);
        }
        // decay mask + bf16 -> ldsS
#pragma unroll
        for (int t2 = 0; t2 < 2; ++t2) {
            const int ji = jbase + t2;
            const int sj = s0 + ji * 16 + (lane & 15);
#pragma unroll
            for (int r = 0; r < 8; ++r) {
                int ml = mi_s * 16 + r + ((lane < 16) ? 0 : 8);
                float dist = (float)(n0 + ml - sj);
                float v = sacc[t2][r];
                v = (dist < 0.0f) ? 0.0f : v * __expf(dist * lg);
                ldsS[ml * RSS + ji * 16 + (lane & 15)] = f2bf(v);
            }
        }
#if HAVE_TDM
        if (wave == 0) {   // V(st) complete (K(st+1) may stay in flight)
            if (st + 1 < nTiles) tdm_wait<1>(); else tdm_wait<0>();
        }
#endif
        __syncthreads();
        // ret += S * V
#pragma unroll
        for (int mi = 0; mi < 4; ++mi) {
#pragma unroll
            for (int kk = 0; kk < 64; kk += 32) {
                v16bf pa = load_frag_row<RSS>(ldsS, mi * 16, kk, lane);
#pragma unroll
                for (int ni = 0; ni < 2; ++ni) {
                    v16bf vf = load_frag_row<RVS>(ldsVt, wave * 32 + ni * 16, kk, lane);
                    acc[mi][ni] = wmma_bf16(pa, vf, acc[mi][ni]);
                }
            }
        }
        kbuf ^= 1;
    }
#pragma unroll
    for (int mi = 0; mi < 4; ++mi)
#pragma unroll
        for (int ni = 0; ni < 2; ++ni) {
            int col = h * HEAD_DIM + wave * 32 + ni * 16 + (lane & 15);
#pragma unroll
            for (int r = 0; r < 8; ++r) {
                int m = mi * 16 + r + ((lane < 16) ? 0 : 8);
                RET[((size_t)(b * SEQ + n0 + m)) * HIDDEN + col] = acc[mi][ni][r];
            }
        }
}

// ---------------------------------------------------------------------------
// Per-head groupnorm + SiLU gate.  One wave32 per (b, n, head).
// ---------------------------------------------------------------------------
__global__ __launch_bounds__(256) void gn_gate_kernel(
    const float* __restrict__ ret, const float* __restrict__ g,
    const float* __restrict__ gnw, const float* __restrict__ gnb,
    __bf16* __restrict__ gate) {
    const int lane = threadIdx.x & 31;
    const int wid = threadIdx.x >> 5;
    const int group = blockIdx.x * 8 + wid;
    const int row = group >> 3, h = group & 7;
    const size_t off = (size_t)row * HIDDEN + (size_t)h * HEAD_DIM;

    float vals[8], sum = 0.0f, sumsq = 0.0f;
#pragma unroll
    for (int i = 0; i < 8; ++i) {
        float v = ret[off + lane + 32 * i];
        vals[i] = v;
        sum += v;
        sumsq += v * v;
    }
#pragma unroll
    for (int o = 16; o > 0; o >>= 1) {
        sum += __shfl_xor(sum, o, 32);
        sumsq += __shfl_xor(sumsq, o, 32);
    }
    const float mean = sum * (1.0f / HEAD_DIM);
    const float var = sumsq * (1.0f / HEAD_DIM) - mean * mean;
    const float inv = rsqrtf(var + 1e-5f);
#pragma unroll
    for (int i = 0; i < 8; ++i) {
        int d = lane + 32 * i;
        float norm = (vals[i] - mean) * inv * gnw[h * HEAD_DIM + d] + gnb[h * HEAD_DIM + d];
        float gv = g[off + d];
        float sil = gv / (1.0f + __expf(-gv));
        gate[off + d] = f2bf(sil * norm);
    }
}

// ---------------------------------------------------------------------------
// Host-side orchestration
// ---------------------------------------------------------------------------
extern "C" void kernel_launch(void* const* d_in, const int* in_sizes, int n_in,
                              void* d_out, int out_size, void* d_ws, size_t ws_size,
                              hipStream_t stream) {
    (void)in_sizes; (void)n_in; (void)out_size; (void)ws_size;
    const float* x    = (const float*)d_in[0];
    const float* w_q  = (const float*)d_in[1];
    const float* b_q  = (const float*)d_in[2];
    const float* w_k  = (const float*)d_in[3];
    const float* b_k  = (const float*)d_in[4];
    const float* w_v  = (const float*)d_in[5];
    const float* b_v  = (const float*)d_in[6];
    const float* w_g  = (const float*)d_in[7];
    const float* b_g  = (const float*)d_in[8];
    const float* w_o  = (const float*)d_in[9];
    const float* b_o  = (const float*)d_in[10];
    const float* gnw  = (const float*)d_in[11];
    const float* gnb  = (const float*)d_in[12];
    float* out = (float*)d_out;

    const size_t SZ_X = (size_t)BN * HIDDEN;
    const size_t SZ_W = (size_t)HIDDEN * HIDDEN;

    char* w = (char*)d_ws;
    __bf16* XB = (__bf16*)w;               w += SZ_X * 2;
    __bf16* WT[5];
    for (int i = 0; i < 5; ++i) { WT[i] = (__bf16*)w; w += SZ_W * 2; }
    __bf16* QBp = (__bf16*)w;              w += SZ_X * 2;
    __bf16* KBp = (__bf16*)w;              w += SZ_X * 2;
    __bf16* VTp = (__bf16*)w;              w += SZ_X * 2;
    float* Q32 = (float*)w;                w += SZ_X * 4;
    float* K32 = (float*)w;                w += SZ_X * 4;
    float* V32 = (float*)w;                w += SZ_X * 4;
    float* G32 = (float*)w;                w += SZ_X * 4;
    float* RET32 = Q32;
    __bf16* GATE = (__bf16*)K32;

    // conversions: x -> bf16; weights -> transposed bf16 (one-time)
    cvt_f32_bf16_kernel<<<4096, 256, 0, stream>>>(x, XB, SZ_X);
    transpose_w_kernel<<<1024, 256, 0, stream>>>(w_q, WT[0]);
    transpose_w_kernel<<<1024, 256, 0, stream>>>(w_k, WT[1]);
    transpose_w_kernel<<<1024, 256, 0, stream>>>(w_v, WT[2]);
    transpose_w_kernel<<<1024, 256, 0, stream>>>(w_g, WT[3]);
    transpose_w_kernel<<<1024, 256, 0, stream>>>(w_o, WT[4]);

    dim3 ggrid(HIDDEN / 128, BN / 128);
    gemm_bf16_kernel<<<ggrid, 256, 0, stream>>>(XB, WT[0], b_q, Q32, BN, HIDDEN, HIDDEN);
    gemm_bf16_kernel<<<ggrid, 256, 0, stream>>>(XB, WT[1], b_k, K32, BN, HIDDEN, HIDDEN);
    gemm_bf16_kernel<<<ggrid, 256, 0, stream>>>(XB, WT[2], b_v, V32, BN, HIDDEN, HIDDEN);
    gemm_bf16_kernel<<<ggrid, 256, 0, stream>>>(XB, WT[3], b_g, G32, BN, HIDDEN, HIDDEN);

    theta_shift_kernel<<<(BN * 1024) / 256, 256, 0, stream>>>(Q32, QBp, 1.0f);
    theta_shift_kernel<<<(BN * 1024) / 256, 256, 0, stream>>>(K32, KBp, 1.0f / 16.0f);
    transpose_v_kernel<<<dim3(SEQ / 64, HEAD_DIM / 64, BATCH * HEADS), 256, 0, stream>>>(V32, VTp);

    dim3 rgrid(SEQ / 64, BATCH * HEADS);
    retention_kernel<<<rgrid, 256, 0, stream>>>(QBp, KBp, VTp, RET32);

    gn_gate_kernel<<<(BN * HEADS) / 8, 256, 0, stream>>>(RET32, G32, gnw, gnb, GATE);

    gemm_bf16_kernel<<<ggrid, 256, 0, stream>>>(GATE, WT[4], b_o, out, BN, HIDDEN, HIDDEN);
}
